// ComboLossV2_3659312136384
// MI455X (gfx1250) — compile-verified, matrix operand order checked by reading, not run on hardware
//
#include <hip/hip_runtime.h>
#include <hip/hip_bf16.h>

typedef __attribute__((ext_vector_type(16))) _Float16 v16h;
typedef __attribute__((ext_vector_type(8)))  float    v8f;
typedef unsigned long long u64;

// accumulator slots (doubles)
#define A_BCE   0
#define A_FOCAL 1
#define A_INTER 2
#define A_PSUM  3
#define A_TSUM  4
#define A_BND   5
#define A_LOV   6
#define A_COUNT 16

#define SORT_CHUNK 4096
#define SCAN_MAXPER 16

__device__ __forceinline__ float wave_red_f32(float v) {
    #pragma unroll
    for (int o = 16; o > 0; o >>= 1) v += __shfl_xor(v, o, 32);
    return v;
}
__device__ __forceinline__ double wave_red_f64(double v) {
    #pragma unroll
    for (int o = 16; o > 0; o >>= 1) v += __shfl_xor(v, o, 32);
    return v;
}

__global__ void init_accum_kernel(double* accum) {
    int t = threadIdx.x;
    if (t < A_COUNT) accum[t] = 0.0;
}

// ---------------------------------------------------------------------------
// Elementwise maps + reductions. Each wave consumes tiles of 512 elements:
// 16 elements/lane -> one v16h operand per quantity -> one
// v_wmma_f32_16x16x32_f16 (B = ones) per quantity per tile. tsum is exact
// (t in {0,1}); psum/boundary accept f16 input rounding. bce/focal/inter are
// accumulated in f32 VALU and co-execute with the XDL WMMAs.
// Also emits the sort items: key = ~bits(|s-t|) (stable descending on error),
// payload g-bit in bit 0 of a u64.
// ---------------------------------------------------------------------------
__global__ void elementwise_kernel(const float* __restrict__ pred,
                                   const float* __restrict__ target,
                                   const float* __restrict__ dist,
                                   u64* __restrict__ items,
                                   double* __restrict__ accum,
                                   long long N, long long numTiles) {
    const int lane = threadIdx.x & 31;
    const long long wid = ((long long)blockIdx.x * blockDim.x + threadIdx.x) >> 5;
    const long long nw  = ((long long)gridDim.x * blockDim.x) >> 5;

    v16h ones;
    #pragma unroll
    for (int j = 0; j < 16; ++j) ones[j] = (_Float16)1.0f;

    v8f accT = {0.f,0.f,0.f,0.f,0.f,0.f,0.f,0.f};
    v8f accP = {0.f,0.f,0.f,0.f,0.f,0.f,0.f,0.f};
    v8f accB = {0.f,0.f,0.f,0.f,0.f,0.f,0.f,0.f};
    float bceA = 0.f, focA = 0.f, intA = 0.f;

    for (long long tile = wid; tile < numTiles; tile += nw) {
        const long long base = tile * 512;
        v16h vt, vp, vb;
        #pragma unroll
        for (int j = 0; j < 16; ++j) {
            const long long idx = base + (long long)j * 32 + lane;
            float ft = 0.f, fs = 0.f, fb = 0.f;
            if (idx < N) {
                const float x  = pred[idx];
                const float t  = target[idx];
                const float dd = dist[idx];
                const float ax = fabsf(x);
                const float bm = fmaxf(x, 0.f) - x * t + log1pf(expf(-ax));
                const float pt = expf(-bm);
                const float om = 1.f - pt;
                const float s  = 1.f / (1.f + expf(-x));
                bceA += bm;
                focA += om * om * bm;
                intA += s * t;
                ft = t;
                fs = s;
                fb = dd * (s - t) * (s - t);
                const float e = fabsf(s - t);
                const unsigned kb = ~__float_as_uint(e);
                items[idx] = ((u64)kb << 32) | (u64)(t > 0.5f ? 1u : 0u);
            }
            vt[j] = (_Float16)ft;
            vp[j] = (_Float16)fs;
            vb[j] = (_Float16)fb;
        }
        accT = __builtin_amdgcn_wmma_f32_16x16x32_f16(false, vt, false, ones,
                                                      (short)0, accT, false, false);
        accP = __builtin_amdgcn_wmma_f32_16x16x32_f16(false, vp, false, ones,
                                                      (short)0, accP, false, false);
        accB = __builtin_amdgcn_wmma_f32_16x16x32_f16(false, vb, false, ones,
                                                      (short)0, accB, false, false);
    }

    // Each 16x16 D replicates row sums over 16 columns -> divide by 16.
    float st = 0.f, sp = 0.f, sb = 0.f;
    #pragma unroll
    for (int j = 0; j < 8; ++j) { st += accT[j]; sp += accP[j]; sb += accB[j]; }

    const float wb = wave_red_f32(bceA);
    const float wf = wave_red_f32(focA);
    const float wi = wave_red_f32(intA);
    const float wt = wave_red_f32(st) * 0.0625f;
    const float wp = wave_red_f32(sp) * 0.0625f;
    const float wbd = wave_red_f32(sb) * 0.0625f;
    if (lane == 0) {
        atomicAdd(&accum[A_BCE],   (double)wb);
        atomicAdd(&accum[A_FOCAL], (double)wf);
        atomicAdd(&accum[A_INTER], (double)wi);
        atomicAdd(&accum[A_TSUM],  (double)wt);
        atomicAdd(&accum[A_PSUM],  (double)wp);
        atomicAdd(&accum[A_BND],   (double)wbd);
    }
}

// ---------------------------------------------------------------------------
// LSD radix sort, 4 passes x 8 bits over bits [39:32]..[63:56] of the u64 item.
// ---------------------------------------------------------------------------
__global__ void radix_hist_kernel(const u64* __restrict__ src,
                                  unsigned* __restrict__ g_hist,
                                  long long N, int numChunks, int shift) {
    __shared__ unsigned h[256];
    const int t = threadIdx.x;
    h[t] = 0u;
    __syncthreads();
    const long long base = (long long)blockIdx.x * SORT_CHUNK;
    #pragma unroll
    for (int k = 0; k < SORT_CHUNK / 256; ++k) {
        const long long i = base + (long long)k * 256 + t;
        if (i < N) {
            const unsigned d = (unsigned)(src[i] >> shift) & 255u;
            atomicAdd(&h[d], 1u);
        }
    }
    __syncthreads();
    g_hist[(long long)t * numChunks + blockIdx.x] = h[t];
}

// One block per row; exclusive scan of `len` entries in place, total -> totals[row].
__global__ void scan_rows_kernel(unsigned* __restrict__ rows,
                                 unsigned* __restrict__ totals, int len) {
    __shared__ unsigned s[256];
    const int t = threadIdx.x;
    unsigned* row = rows + (long long)blockIdx.x * len;
    const int per = (len + 255) / 256;
    unsigned local[SCAN_MAXPER];
    unsigned sum = 0u;
    for (int k = 0; k < per; ++k) {
        const int c = t * per + k;
        const unsigned v = (c < len) ? row[c] : 0u;
        local[k] = v;
        sum += v;
    }
    s[t] = sum;
    __syncthreads();
    #pragma unroll
    for (int off = 1; off < 256; off <<= 1) {
        const unsigned v = (t >= off) ? s[t - off] : 0u;
        __syncthreads();
        s[t] += v;
        __syncthreads();
    }
    unsigned run = s[t] - sum;  // exclusive prefix for this thread's span
    for (int k = 0; k < per; ++k) {
        const int c = t * per + k;
        if (c < len) { row[c] = run; run += local[k]; }
    }
    if (t == 255) totals[blockIdx.x] = s[255];
}

__global__ void binbase_kernel(const unsigned* __restrict__ digit_total,
                               unsigned* __restrict__ binbase) {
    __shared__ unsigned s[256];
    const int t = threadIdx.x;
    const unsigned v = digit_total[t];
    s[t] = v;
    __syncthreads();
    #pragma unroll
    for (int off = 1; off < 256; off <<= 1) {
        const unsigned x = (t >= off) ? s[t - off] : 0u;
        __syncthreads();
        s[t] += x;
        __syncthreads();
    }
    binbase[t] = s[t] - v;  // exclusive
}

// One wave per chunk; stable in-wave ranking via 8 ballots per batch of 32.
__global__ void radix_scatter_kernel(const u64* __restrict__ src,
                                     u64* __restrict__ dst,
                                     const unsigned* __restrict__ g_hist,
                                     const unsigned* __restrict__ binbase,
                                     long long N, int numChunks, int shift) {
    __shared__ unsigned counters[256];
    __shared__ unsigned basepref[256];
    const int lane = threadIdx.x;  // blockDim.x == 32
    const int chunk = blockIdx.x;
    #pragma unroll
    for (int k = 0; k < 8; ++k) {
        const int d = lane + k * 32;
        counters[d] = 0u;
        basepref[d] = binbase[d] + g_hist[(long long)d * numChunks + chunk];
    }
    __syncthreads();
    const long long base = (long long)chunk * SORT_CHUNK;
    const unsigned lt = (lane == 0) ? 0u : (0xFFFFFFFFu >> (32 - lane));
    for (int b = 0; b < SORT_CHUNK / 32; ++b) {
        const long long i = base + (long long)b * 32 + lane;
        const bool valid = (i < N);
        const u64 item = valid ? src[i] : 0ull;
        const unsigned dig = (unsigned)(item >> shift) & 255u;
        unsigned m = (unsigned)__ballot(valid ? 1 : 0);
        #pragma unroll
        for (int bit = 0; bit < 8; ++bit) {
            const unsigned vote = (unsigned)__ballot((valid && ((dig >> bit) & 1u)) ? 1 : 0);
            m &= ((dig >> bit) & 1u) ? vote : ~vote;
        }
        const unsigned earlier = __popc(m & lt);
        const unsigned cnt = __popc(m);
        if (valid) {
            const unsigned pos = basepref[dig] + counters[dig] + earlier;
            dst[pos] = item;
        }
        __syncthreads();
        if (valid && earlier == cnt - 1u) counters[dig] += cnt;  // group leader
        __syncthreads();
    }
}

// ---------------------------------------------------------------------------
// Prefix sum of sorted g-bits + Lovász gradient dot product.
// ---------------------------------------------------------------------------
__global__ void sumg_kernel(const u64* __restrict__ items,
                            unsigned* __restrict__ blocksum, long long N) {
    __shared__ unsigned ws[8];
    const int t = threadIdx.x;
    const long long base = (long long)blockIdx.x * SORT_CHUNK;
    unsigned s = 0u;
    #pragma unroll
    for (int k = 0; k < SORT_CHUNK / 256; ++k) {
        const long long i = base + (long long)k * 256 + t;
        if (i < N) s += (unsigned)(items[i] & 1ull);
    }
    #pragma unroll
    for (int o = 16; o > 0; o >>= 1) s += __shfl_xor(s, o, 32);
    if ((t & 31) == 0) ws[t >> 5] = s;
    __syncthreads();
    if (t == 0) {
        unsigned tot = 0u;
        #pragma unroll
        for (int w = 0; w < 8; ++w) tot += ws[w];
        blocksum[blockIdx.x] = tot;
    }
}

__global__ void lovasz_kernel(const u64* __restrict__ items,
                              const unsigned* __restrict__ blockexcl,
                              double* __restrict__ accum, long long N) {
    __shared__ unsigned sh[256];
    __shared__ double wd[8];
    const int t = threadIdx.x;
    const long long base = (long long)blockIdx.x * SORT_CHUNK;
    const double gts = accum[A_TSUM];

    unsigned gloc[16];
    float eloc[16];
    unsigned s = 0u;
    #pragma unroll
    for (int k = 0; k < 16; ++k) {
        const long long i = base + (long long)t * 16 + k;
        unsigned g = 0u; float e = 0.f;
        if (i < N) {
            const u64 it = items[i];
            g = (unsigned)(it & 1ull);
            e = __uint_as_float(~(unsigned)(it >> 32));
        }
        gloc[k] = g; eloc[k] = e; s += g;
    }
    sh[t] = s;
    __syncthreads();
    #pragma unroll
    for (int off = 1; off < 256; off <<= 1) {
        const unsigned v = (t >= off) ? sh[t - off] : 0u;
        __syncthreads();
        sh[t] += v;
        __syncthreads();
    }
    unsigned run = sh[t] - s + blockexcl[blockIdx.x];  // exclusive prefix
    double part = 0.0;
    #pragma unroll
    for (int k = 0; k < 16; ++k) {
        const long long i = base + (long long)t * 16 + k;
        if (i < N) {
            run += gloc[k];
            const double cs = (double)run;
            const double jac = 1.0 - (gts - cs) / (gts + (double)(i + 1) - cs);
            double grad;
            if (i == 0) {
                grad = jac;
            } else {
                const double csm = cs - (double)gloc[k];
                const double jm = 1.0 - (gts - csm) / (gts + (double)i - csm);
                grad = jac - jm;
            }
            part += (double)eloc[k] * grad;
        }
    }
    part = wave_red_f64(part);
    if ((t & 31) == 0) wd[t >> 5] = part;
    __syncthreads();
    if (t == 0) {
        double tot = 0.0;
        #pragma unroll
        for (int w = 0; w < 8; ++w) tot += wd[w];
        atomicAdd(&accum[A_LOV], tot);
    }
}

__global__ void finalize_kernel(const double* __restrict__ accum,
                                float* __restrict__ out, double N) {
    if (threadIdx.x == 0 && blockIdx.x == 0) {
        const double SMOOTH = 1e-6;
        const double bce   = accum[A_BCE] / N;
        const double focal = accum[A_FOCAL] / N;
        const double inter = accum[A_INTER];
        const double psum  = accum[A_PSUM];
        const double tsum  = accum[A_TSUM];
        const double bnd   = accum[A_BND] / N;
        const double lov   = accum[A_LOV];
        const double dice = 1.0 - (2.0 * inter + SMOOTH) / (psum + tsum + SMOOTH);
        const double fp = psum - inter;
        const double fn = tsum - inter;
        const double tver = 1.0 - (inter + SMOOTH) / (inter + 0.7 * fp + 0.3 * fn + SMOOTH);
        const double w_bce = 1.0 * bce;
        const double w_dce = 1.0 * dice;
        const double w_foc = 1.0 * focal;
        const double w_tv  = 0.5 * tver;
        const double w_bnd = 0.3 * bnd;
        const double w_lov = 0.2 * lov;
        const double total = w_bce + w_dce + w_foc + w_tv + w_bnd + w_lov;
        out[0] = (float)total;
        out[1] = (float)w_bce;
        out[2] = (float)w_dce;
        out[3] = (float)w_foc;
        out[4] = (float)w_tv;
        out[5] = (float)w_bnd;
        out[6] = (float)w_lov;
    }
}

extern "C" void kernel_launch(void* const* d_in, const int* in_sizes, int n_in,
                              void* d_out, int out_size, void* d_ws, size_t ws_size,
                              hipStream_t stream) {
    const float* pred   = (const float*)d_in[0];
    const float* target = (const float*)d_in[1];
    const float* dist   = (const float*)d_in[2];
    float* out = (float*)d_out;
    const long long N = (long long)in_sizes[0];
    const int numChunks = (int)((N + SORT_CHUNK - 1) / SORT_CHUNK);

    // workspace carve-up
    char* ws = (char*)d_ws;
    size_t off = 0;
    auto carve = [&](size_t bytes) -> void* {
        off = (off + 255) & ~(size_t)255;
        void* p = ws + off;
        off += bytes;
        return p;
    };
    u64* bufA            = (u64*)carve((size_t)N * 8);
    u64* bufB            = (u64*)carve((size_t)N * 8);
    unsigned* g_hist     = (unsigned*)carve((size_t)256 * numChunks * 4);
    unsigned* g_dtotal   = (unsigned*)carve(256 * 4);
    unsigned* g_binbase  = (unsigned*)carve(256 * 4);
    unsigned* g_blocksum = (unsigned*)carve((size_t)numChunks * 4);
    double* accum        = (double*)carve(A_COUNT * 8);
    (void)ws_size;

    init_accum_kernel<<<1, 32, 0, stream>>>(accum);

    const long long numTiles = (N + 511) / 512;
    int ewBlocks = (int)((numTiles + 7) / 8);
    if (ewBlocks > 4096) ewBlocks = 4096;
    if (ewBlocks < 1) ewBlocks = 1;
    elementwise_kernel<<<ewBlocks, 256, 0, stream>>>(pred, target, dist, bufA,
                                                     accum, N, numTiles);

    u64* src = bufA;
    u64* dst = bufB;
    for (int pass = 0; pass < 4; ++pass) {
        const int shift = 32 + 8 * pass;
        radix_hist_kernel<<<numChunks, 256, 0, stream>>>(src, g_hist, N, numChunks, shift);
        scan_rows_kernel<<<256, 256, 0, stream>>>(g_hist, g_dtotal, numChunks);
        binbase_kernel<<<1, 256, 0, stream>>>(g_dtotal, g_binbase);
        radix_scatter_kernel<<<numChunks, 32, 0, stream>>>(src, dst, g_hist, g_binbase,
                                                           N, numChunks, shift);
        u64* tmp = src; src = dst; dst = tmp;
    }
    // 4 passes: sorted data back in bufA (== src)

    sumg_kernel<<<numChunks, 256, 0, stream>>>(src, g_blocksum, N);
    scan_rows_kernel<<<1, 256, 0, stream>>>(g_blocksum, g_dtotal, numChunks);
    lovasz_kernel<<<numChunks, 256, 0, stream>>>(src, g_blocksum, accum, N);

    finalize_kernel<<<1, 32, 0, stream>>>(accum, out, (double)N);
}
